// SplatImageAttention_2190433320968
// MI455X (gfx1250) — matrix operand order repaired and checked by reading, not compiled, not run on hardware
//
#include <hip/hip_runtime.h>

typedef _Float16 half4 __attribute__((ext_vector_type(4)));
typedef _Float16 half8 __attribute__((ext_vector_type(8)));
typedef _Float16 v16h  __attribute__((ext_vector_type(16)));
typedef float    v8f   __attribute__((ext_vector_type(8)));

#define T_TOK   1024          // image/block tokens
#define C_DIM   64            // feature dim
#define ROWPAD  72            // padded LDS row length in f16 (bank-conflict-free)
#define NWAVE   8             // waves per block (wave32 -> 256 threads)
#define TPW     8             // i-tiles (of 16 rows) per wave; 8*8*16 = 1024 rows

// Single-instruction max: avoids llvm.maxnum sNaN canonicalization (extra
// self-max ops) and cmp/cndmask lowering (vcc-serialized 2-op chains).
__device__ __forceinline__ float fmax_asm(float a, float b) {
    float r;
    asm("v_max_num_f32 %0, %1, %2" : "=v"(r) : "v"(a), "v"(b));
    return r;
}

// One block per layer l. Stages Q[l] and K into LDS as f16 (padded rows),
// then each wave computes a 128-row strip of scores = Q K^T with
// v_wmma_f32_16x16x32_f16, tracking per-row running max, and the block
// writes sum-of-row-maxes to out[l]. No atomics -> deterministic.
extern "C" __global__ __launch_bounds__(NWAVE * 32, 1)
void splat_attn_wmma(const float* __restrict__ Q,
                     const float* __restrict__ K,
                     float* __restrict__ out)
{
    extern __shared__ _Float16 smem[];           // dynamic LDS, 2 * 1024 * 72 f16
    _Float16* Qs = smem;                          // [1024][ROWPAD]
    _Float16* Ks = smem + T_TOK * ROWPAD;         // [1024][ROWPAD]
    __shared__ float wsum[NWAVE];

    const int l    = blockIdx.x;
    const int tid  = threadIdx.x;
    const int wave = tid >> 5;
    const int lane = tid & 31;

    // ---------------- Stage Q[l] and K into LDS as f16 ----------------
    const float4* Ql4 = (const float4*)(Q + (size_t)l * (T_TOK * C_DIM));
    const float4* K4  = (const float4*)K;
    for (int idx4 = tid; idx4 < (T_TOK * C_DIM) / 4; idx4 += NWAVE * 32) {
        const int row = idx4 >> 4;               // 16 float4 per 64-wide row
        const int col = (idx4 & 15) << 2;
        float4 q4 = Ql4[idx4];
        float4 k4 = K4[idx4];
        half4 qh = { (_Float16)q4.x, (_Float16)q4.y, (_Float16)q4.z, (_Float16)q4.w };
        half4 kh = { (_Float16)k4.x, (_Float16)k4.y, (_Float16)k4.z, (_Float16)k4.w };
        *(half4*)(&Qs[row * ROWPAD + col]) = qh;  // ds_store_b64, 8B aligned
        *(half4*)(&Ks[row * ROWPAD + col]) = kh;
    }
    __syncthreads();

    // ---------------- Load resident A fragments (Q tiles) ----------------
    // A 16x32 f16 layout (ISA 7.12.2): lane (m = lane&15, h = lane>>4) holds
    // K = [8h, 8h+8) in v16h[0..7] and K = [16+8h, 16+8h+8) in v16h[8..15].
    const int m = lane & 15;
    const int h = lane >> 4;

    v16h A[TPW][2];
    #pragma unroll
    for (int t = 0; t < TPW; ++t) {
        const int row = (wave * TPW + t) * 16 + m;
        #pragma unroll
        for (int s = 0; s < 2; ++s) {             // k-step: channels [32s, 32s+32)
            const _Float16* p = &Qs[row * ROWPAD + 32 * s + 8 * h];
            half8 lo = *(const half8*)p;          // ds_load_b128 (16B aligned)
            half8 hi = *(const half8*)(p + 16);
            A[t][s] = __builtin_shufflevector(lo, hi,
                        0,1,2,3,4,5,6,7, 8,9,10,11,12,13,14,15);
        }
    }

    v8f vmax[TPW];
    #pragma unroll
    for (int t = 0; t < TPW; ++t)
        #pragma unroll
        for (int r = 0; r < 8; ++r)
            vmax[t][r] = -__builtin_huge_valf();

    // ---------------- j loop over 64 column tiles of 16 tokens ----------------
    // B 32x16 f16 layout: lane (n = lane&15, h) holds K = [8h, 8h+8) and [16+8h, 16+8h+8).
    // Software-pipelined: ping-pong B buffers, loads for j+1 issued before the
    // WMMAs for j so the ds_load latency is hidden under 16 WMMAs.
    const _Float16* bbase = &Ks[(lane & 15) * ROWPAD + 8 * h];

    #define LOAD_B(Bv, jt)                                                    \
        do {                                                                  \
            _Pragma("unroll")                                                 \
            for (int s = 0; s < 2; ++s) {                                     \
                const _Float16* p = bbase + (jt) * 16 * ROWPAD + 32 * s;      \
                half8 lo = *(const half8*)p;                                  \
                half8 hi = *(const half8*)(p + 16);                           \
                Bv[s] = __builtin_shufflevector(lo, hi,                       \
                            0,1,2,3,4,5,6,7, 8,9,10,11,12,13,14,15);          \
            }                                                                 \
        } while (0)

    #define COMPUTE(Bv)                                                       \
        do {                                                                  \
            _Pragma("unroll")                                                 \
            for (int t = 0; t < TPW; ++t) {                                   \
                v8f d = {};                                                   \
                d = __builtin_amdgcn_wmma_f32_16x16x32_f16(false, A[t][0],    \
                        false, Bv[0], (short)0, d, false, false);             \
                d = __builtin_amdgcn_wmma_f32_16x16x32_f16(false, A[t][1],    \
                        false, Bv[1], (short)0, d, false, false);             \
                _Pragma("unroll")                                             \
                for (int r = 0; r < 8; ++r)                                   \
                    vmax[t][r] = fmax_asm(vmax[t][r], d[r]);                  \
            }                                                                 \
        } while (0)

    v16h B0[2], B1[2];
    LOAD_B(B0, 0);
    for (int jt = 0; jt < T_TOK / 16; jt += 2) {
        LOAD_B(B1, jt + 1);                       // prefetch odd tile
        COMPUTE(B0);                              // compute even tile
        LOAD_B(B0, (jt + 2) & 63);                // prefetch next even (wraps, harmless)
        COMPUTE(B1);                              // compute odd tile
    }
    #undef LOAD_B
    #undef COMPUTE

    // ---------------- Reduce: per-row max across 16 lanes, then sum ----------------
    // D layout: v8f[r] holds row M = r + 8h, column N = lane&15.
    float s_local = 0.0f;
    #pragma unroll
    for (int t = 0; t < TPW; ++t) {
        #pragma unroll
        for (int r = 0; r < 8; ++r) {
            float v = vmax[t][r];
            v = fmax_asm(v, __shfl_xor(v, 1, 32));
            v = fmax_asm(v, __shfl_xor(v, 2, 32));
            v = fmax_asm(v, __shfl_xor(v, 4, 32));
            v = fmax_asm(v, __shfl_xor(v, 8, 32));
            s_local += v;                          // row max of row r+8h of tile t
        }
    }
    s_local += __shfl_xor(s_local, 16, 32);        // combine the two 16-lane halves
    if (lane == 0) wsum[wave] = s_local;
    __syncthreads();

    if (tid == 0) {
        float acc = 0.0f;
        #pragma unroll
        for (int w = 0; w < NWAVE; ++w) acc += wsum[w];
        out[l] = acc;
    }
}

extern "C" void kernel_launch(void* const* d_in, const int* in_sizes, int n_in,
                              void* d_out, int out_size, void* d_ws, size_t ws_size,
                              hipStream_t stream) {
    (void)n_in; (void)d_ws; (void)ws_size; (void)in_sizes;
    const float* Q = (const float*)d_in[0];       // [L, T, C] f32
    const float* K = (const float*)d_in[1];       // [T, C] f32
    float* out = (float*)d_out;                   // [L] f32

    const int L = out_size;                       // 256
    const size_t smem = (size_t)2 * T_TOK * ROWPAD * sizeof(_Float16); // 288 KB (<320 KB WGP LDS)
    hipFuncSetAttribute((const void*)splat_attn_wmma,
                        hipFuncAttributeMaxDynamicSharedMemorySize, (int)smem);
    splat_attn_wmma<<<L, NWAVE * 32, smem, stream>>>(Q, K, out);
}
